// FeedBack_41205916238533
// MI455X (gfx1250) — matrix operand order repaired and checked by reading, not compiled, not run on hardware
//
#include <hip/hip_runtime.h>
#include <hip/hip_bf16.h>

typedef __attribute__((ext_vector_type(16))) _Float16 v16h;
typedef __attribute__((ext_vector_type(8)))  float    v8f;
typedef __attribute__((ext_vector_type(4)))  float    v4f;   // builtin vec: AS-cast friendly

// Explicit global-address-space views so laundered pointers still lower to
// global_load_b128 (LOADcnt only) instead of flat_load_b128 (LOADcnt+DScnt).
typedef __attribute__((address_space(1))) const _Float16 gf16;
typedef __attribute__((address_space(1))) const v4f      gv4f;

#define T_STEPS 2048
#define UNITS   256
#define GDIM    1024   // 4*UNITS
#define HPAD    264    // 256 + 8 halfs padding -> 528B row stride (16B aligned, bank-friendly)

union frag_t { v16h v; v4f q[2]; };

// ---------------------------------------------------------------------------
// Prep: Ut[col][k] = (f16) U[k][col]   (U is [256][1024] f32, row-major)
// Ut is [1024][256] f16 in workspace -> B fragments become contiguous loads.
// ---------------------------------------------------------------------------
__global__ void transpose_u_kernel(const float* __restrict__ U,
                                   _Float16* __restrict__ Ut) {
    int idx = blockIdx.x * blockDim.x + threadIdx.x;   // 0 .. 1024*256-1
    if (idx >= GDIM * UNITS) return;
    int col = idx >> 8;        // 0..1023
    int k   = idx & 255;       // 0..255
    Ut[col * UNITS + k] = (_Float16)U[k * GDIM + col];
}

// ---------------------------------------------------------------------------
// Persistent per-batch-tile LSTM scan.
// grid = 8 blocks (16 batch rows each), block = 512 threads = 16 wave32s.
// Wave w owns unit tile [16w, 16w+16); keeps its c-state tile in VGPRs.
// ---------------------------------------------------------------------------
__global__ __launch_bounds__(512, 1)
void lstm_scan_kernel(const float* __restrict__ x,     // [128][2048][3]
                      const float* __restrict__ W,     // [3][1024]
                      const float* __restrict__ b,     // [1024]
                      const _Float16* __restrict__ Ut, // [1024][256] f16
                      const float* __restrict__ Wd,    // [256][3]
                      const float* __restrict__ bd,    // [3]
                      float* __restrict__ out)         // [128][3]
{
    __shared__ _Float16 hbuf[2][16][HPAD];  // double-buffered hidden state (f16)
    __shared__ float    xs[2][16][4];       // double-buffered x_t slice

    const int tid     = threadIdx.x;
    const int lane    = tid & 31;
    const int wave    = tid >> 5;          // 0..15  -> unit tile
    const int ubase   = wave * 16;
    const int nloc    = lane & 15;         // N (column) within tile
    const int hiHalf  = (lane >> 4) & 1;   // lane 16..31?
    const int mrow0   = hiHalf ? 8 : 0;    // C-layout: VGPR j holds M=j (+8 for hi half)
    const int bofs    = blockIdx.x * 16;   // first batch row of this block

    // Zero both h buffers (t=0 reads zeros from buffer 0).
    for (int i = tid; i < 2 * 16 * HPAD; i += 512)
        ((_Float16*)hbuf)[i] = (_Float16)0.0f;

    // Stage x for t = 0 into xs[0].
    if (tid < 48) {
        int r = tid / 3, d = tid % 3;
        xs[0][r][d] = x[((bofs + r) * T_STEPS + 0) * 3 + d];
    }

    // Hoist per-lane input-projection constants: W[:,col], b[col] per gate.
    float wg0[4], wg1[4], wg2[4], bg[4];
#pragma unroll
    for (int g = 0; g < 4; ++g) {
        int col = g * UNITS + ubase + nloc;
        wg0[g] = W[0 * GDIM + col];
        wg1[g] = W[1 * GDIM + col];
        wg2[g] = W[2 * GDIM + col];
        bg[g]  = b[col];
    }

    v8f c = {};   // cell state tile, f32, lives in VGPRs for the whole scan

    __syncthreads();

    const int acol0 = hiHalf ? 8 : 0;   // A-fragment K offset for this lane half
    const int arow  = nloc;             // A-fragment row M = lane%16
    const int bkoff = hiHalf ? 8 : 0;   // B-fragment K offset for this lane half
    const int boff0 = (ubase + nloc) * UNITS + bkoff;  // per-lane B element offset

    for (int t = 0; t < T_STEPS; ++t) {
        const int rb = t & 1;
        const int wb = rb ^ 1;

        // Launder the U base pointer each timestep so the compiler cannot
        // hoist the (t-invariant) B-fragment loads out of the scan loop —
        // hoisting 256 VGPRs of U spills to scratch, which is far worse than
        // re-streaming from the 192MB L2 (Ut is only 512KB, L2-resident).
        const _Float16* ut_t = Ut;
        asm volatile("" : "+s"(ut_t));
        const gf16* ubp = (const gf16*)ut_t + boff0;   // global AS -> global_load

        // --- z accumulators initialized with xz = x_t @ W + b (C layout) ---
        v8f acc[4];
#pragma unroll
        for (int g = 0; g < 4; ++g) {
#pragma unroll
            for (int j = 0; j < 8; ++j) {
                int r = mrow0 + j;
                acc[g][j] = bg[g]
                          + xs[rb][r][0] * wg0[g]
                          + xs[rb][r][1] * wg1[g]
                          + xs[rb][r][2] * wg2[g];
            }
        }

        // --- stage x for step t+1 (consumed after this step's barrier) ---
        if (tid < 48 && (t + 1) < T_STEPS) {
            int r = tid / 3, d = tid % 3;
            xs[wb][r][d] = x[((bofs + r) * T_STEPS + (t + 1)) * 3 + d];
        }

        // --- z += h @ U over K = 256 in 8 chunks of 32, double-buffered ----
        frag_t Af[2];
        frag_t Bf[2][4];
        {   // preload chunk 0
            const _Float16* hp = &hbuf[rb][arow][acol0];
            Af[0].q[0] = *(const v4f*)hp;
            Af[0].q[1] = *(const v4f*)(hp + 16);
#pragma unroll
            for (int g = 0; g < 4; ++g) {
                const gf16* up = ubp + g * (UNITS * UNITS);
                Bf[0][g].q[0] = *(const gv4f*)up;
                Bf[0][g].q[1] = *(const gv4f*)(up + 16);
            }
        }
#pragma unroll
        for (int kk = 0; kk < 8; ++kk) {
            const int cur = kk & 1;
            const int nxt = cur ^ 1;
            if (kk < 7) {   // issue next chunk's loads under current WMMAs
                const _Float16* hp = &hbuf[rb][arow][(kk + 1) * 32 + acol0];
                Af[nxt].q[0] = *(const v4f*)hp;
                Af[nxt].q[1] = *(const v4f*)(hp + 16);
#pragma unroll
                for (int g = 0; g < 4; ++g) {
                    const gf16* up = ubp + g * (UNITS * UNITS) + (kk + 1) * 32;
                    Bf[nxt][g].q[0] = *(const gv4f*)up;
                    Bf[nxt][g].q[1] = *(const gv4f*)(up + 16);
                }
            }
#pragma unroll
            for (int g = 0; g < 4; ++g) {
                acc[g] = __builtin_amdgcn_wmma_f32_16x16x32_f16(
                    false, Af[cur].v, false, Bf[cur][g].v,
                    (short)0, acc[g], false, false);
            }
        }

        // --- gate math (Keras order i, f, g, o), c in registers ---
#pragma unroll
        for (int j = 0; j < 8; ++j) {
            float zi = acc[0][j], zf = acc[1][j], zg = acc[2][j], zo = acc[3][j];
            float ig = 1.0f / (1.0f + __expf(-zi));
            float fg = 1.0f / (1.0f + __expf(-zf));
            float gg = fmaxf(zg, 0.0f);            // activation = relu
            float cn = fg * c[j] + ig * gg;
            c[j] = cn;
            float og = 1.0f / (1.0f + __expf(-zo));
            float hv = og * fmaxf(cn, 0.0f);
            hbuf[wb][mrow0 + j][ubase + nloc] = (_Float16)hv;
        }

        __syncthreads();   // all h writes done before next step's A reads
    }

    // --- final Dense(3): h_last is in hbuf[0] (T even) ---
    if (wave < 3 && lane < 16) {
        int r = lane, jj = wave;
        float s = bd[jj];
        for (int u = 0; u < UNITS; ++u)
            s += (float)hbuf[0][r][u] * Wd[u * 3 + jj];
        out[(bofs + r) * 3 + jj] = s;
    }
}

// ---------------------------------------------------------------------------
extern "C" void kernel_launch(void* const* d_in, const int* in_sizes, int n_in,
                              void* d_out, int out_size, void* d_ws, size_t ws_size,
                              hipStream_t stream) {
    const float* x  = (const float*)d_in[0];   // [128,2048,3]
    const float* W  = (const float*)d_in[1];   // [3,1024]
    const float* U  = (const float*)d_in[2];   // [256,1024]
    const float* b  = (const float*)d_in[3];   // [1024]
    const float* Wd = (const float*)d_in[4];   // [256,3]
    const float* bd = (const float*)d_in[5];   // [3]
    float* out = (float*)d_out;                // [128,3]

    _Float16* Ut = (_Float16*)d_ws;            // 1024*256*2 = 512 KB

    // Prep: transpose + fp16-convert U (L2-resident thereafter).
    transpose_u_kernel<<<(GDIM * UNITS + 255) / 256, 256, 0, stream>>>(U, Ut);

    // Persistent scan: 8 batch tiles x 16 waves, WMMA per step.
    lstm_scan_kernel<<<8, 512, 0, stream>>>(x, W, b, Ut, Wd, bd, out);
}
